// MoEModel_76020921139218
// MI455X (gfx1250) — compile-verified, matrix operand order checked by reading, not run on hardware
//
#include <hip/hip_runtime.h>
#include <cstdint>

// ---------------------------------------------------------------------------
// MoE fused forward for MI455X (gfx1250, wave32, WMMA 16x16x32 bf16).
//
// Pipeline (all on `stream`):
//   1. convert_x      : x fp32 -> Xb bf16 [N,D] row-major (vectorized)
//   2. pack_transpose : W1 [E,D,H] -> W1p [E,H,D] bf16 (LDS tiled transpose)
//                       W2 [E,H,G] -> W2p [E,G,H] bf16
//   3. gating         : top-2 mask + renorm -> g (d_out[N..])
//   4. fused_moe      : per (64-token tile, expert):
//                         h1 = relu(X@W1+b1) -> LDS bf16 (64 x 2048, padded)
//                         h2 = relu(h1@W2+b2) kept in REGISTERS, folded
//                         directly into y += h2 . w3 via ds_add_f32 reduce
//                         y -> sigmoid -> ybuf [E,N] in ws
//   5. combine        : pred[n] = sum_e g[n,e]*ybuf[e,n] -> d_out[0..N)
//
// Register blocking: each wave runs 4 M-blocks x 2 N-blocks per K-step:
//   8 v_wmma per 12 global_load_b128 (A reused across N, B across M).
//
// ws layout (~72.1 MB): Xb 8MB | W1p 32MB | W2p 32MB | ybuf 128KB
//
// WMMA fragment layout (CDNA5 ISA 7.12.2):
//   A (16x32 bf16): lane L<16 -> row L, K {0..7,16..23}; lane L+16 -> row L,
//     K {8..15,24..31}; two contiguous b128 loads per lane.
//   B (32x16 bf16): assumed mirror per *column* (weights pre-transposed).
//   C/D (16x16 f32): elem r <-> (M = r + 8*(lane>>4), N = lane&15).
// ---------------------------------------------------------------------------

#define N_TOK 4096
#define D_IN  1024
#define H_DIM 2048
#define G_DIM 1024
#define N_EXP 8
#define EPS_G 1e-10f

#define MT    64                       // token tile per workgroup
#define H_LD  (H_DIM + 8)              // padded LDS stride (bf16)

#define H1_BYTES   (MT * H_LD * 2)     // 263,168
#define YACC_BYTES (MT * 4)            // 256
#define SMEM_BYTES (H1_BYTES + YACC_BYTES)   // ~257 KB < 320 KB WGP LDS

typedef __attribute__((ext_vector_type(16))) __bf16 v16bf;
typedef __attribute__((ext_vector_type(8)))  float  v8f;
typedef __attribute__((ext_vector_type(4)))  int    i4v;
typedef __attribute__((ext_vector_type(4)))  float  f4v;
typedef __attribute__((ext_vector_type(4)))  __bf16 bf4v;

// One 16x32 bf16 fragment per lane: 8 contiguous bf16 at p (K+0..7) and 8 at
// p+16 (K+16..23) -> two b128 loads.
__device__ __forceinline__ v16bf load_frag(const __bf16* p) {
    union { i4v q[2]; v16bf v; } u;
    u.q[0] = *(const i4v*)(p);
    u.q[1] = *(const i4v*)(p + 16);
    return u.v;
}

// ---------------------------------------------------------------- pre-passes
__global__ void convert_x_kernel(const float* __restrict__ x,
                                 __bf16* __restrict__ xb) {
    int i = blockIdx.x * blockDim.x + threadIdx.x;   // N*D/4 threads
    f4v v = ((const f4v*)x)[i];
    bf4v o;
#pragma unroll
    for (int j = 0; j < 4; ++j) o[j] = (__bf16)v[j];
    ((bf4v*)xb)[i] = o;
}

// Per-expert 2D transpose with fp32->bf16 convert: in [E][R][C] -> out
// [E][C][R]. LDS 32x32 tiles, coalesced reads AND writes.
// grid: (C/32, R/32, E), block: (32, 8).
__global__ void pack_transpose_kernel(const float* __restrict__ in,
                                      __bf16* __restrict__ out,
                                      int R, int C) {
    __shared__ float tile[32][33];
    const int e  = blockIdx.z;
    const int c0 = blockIdx.x * 32;
    const int r0 = blockIdx.y * 32;
    const float* ine  = in  + (size_t)e * R * C;
    __bf16*      oute = out + (size_t)e * R * C;
    const int tx = threadIdx.x, ty = threadIdx.y;
#pragma unroll
    for (int i = 0; i < 4; ++i)
        tile[ty + i * 8][tx] = ine[(size_t)(r0 + ty + i * 8) * C + (c0 + tx)];
    __syncthreads();
#pragma unroll
    for (int i = 0; i < 4; ++i)
        oute[(size_t)(c0 + ty + i * 8) * R + (r0 + tx)] =
            (__bf16)tile[tx][ty + i * 8];
}

// --------------------------------------------------------------- gating
__global__ void gating_kernel(const float* __restrict__ gates,
                              float* __restrict__ gout) {
    int n = blockIdx.x * blockDim.x + threadIdx.x;
    if (n >= N_TOK) return;
    float v[N_EXP];
#pragma unroll
    for (int e = 0; e < N_EXP; ++e) v[e] = gates[n * N_EXP + e];
    int i1 = 0;
#pragma unroll
    for (int e = 1; e < N_EXP; ++e) if (v[e] > v[i1]) i1 = e;
    int i2 = (i1 == 0) ? 1 : 0;
#pragma unroll
    for (int e = 0; e < N_EXP; ++e) if (e != i1 && v[e] > v[i2]) i2 = e;
    float s = v[i1] + v[i2] + EPS_G;
#pragma unroll
    for (int e = 0; e < N_EXP; ++e)
        gout[n * N_EXP + e] = (e == i1 || e == i2) ? v[e] / s : 0.0f;
}

// --------------------------------------------------------------- fused MLP
// grid: (N/64, E). block: 256 threads = 8 waves (wave32).
__global__ void __launch_bounds__(256)
fused_moe_kernel(const __bf16* __restrict__ Xb,
                 const __bf16* __restrict__ W1p,
                 const __bf16* __restrict__ W2p,
                 const float* __restrict__ b1,
                 const float* __restrict__ b2,
                 const float* __restrict__ W3,
                 const float* __restrict__ b3,
                 float* __restrict__ ybuf) {
    extern __shared__ char smem[];
    __bf16* h1s   = (__bf16*)smem;                 // [MT][H_LD] bf16
    float*  y_acc = (float*)(smem + H1_BYTES);     // [MT]

    const int m0   = blockIdx.x * MT;
    const int e    = blockIdx.y;
    const int lane = threadIdx.x & 31;
    const int wv   = threadIdx.x >> 5;             // 0..7
    const int lo   = lane & 15;
    const int hi   = lane >> 4;

    if (threadIdx.x < MT) y_acc[threadIdx.x] = 0.0f;

    // ---- Phase A: h1 = relu(Xtile @ W1[e] + b1[e]) -> LDS bf16 ----
    // 4 Mblk x 128 Nblk; wave owns 16 N-cols; 4M x 2N register blocking.
    {
        const __bf16* w1e = W1p + (size_t)e * H_DIM * D_IN;
        const __bf16* arow[4];
#pragma unroll
        for (int mb = 0; mb < 4; ++mb)
            arow[mb] = Xb + (size_t)(m0 + mb * 16 + lo) * D_IN;

        for (int ng = 0; ng < 8; ++ng) {           // N-groups of 2
            const int nbase = wv * 16 + ng * 2;
            const __bf16* bcol[2];
#pragma unroll
            for (int j = 0; j < 2; ++j)
                bcol[j] = w1e + (size_t)((nbase + j) * 16 + lo) * D_IN;

            v8f acc[4][2];
#pragma unroll
            for (int mb = 0; mb < 4; ++mb)
#pragma unroll
                for (int j = 0; j < 2; ++j) acc[mb][j] = (v8f){};

            for (int kc = 0; kc < D_IN / 32; ++kc) {
                const int kof = kc * 32 + hi * 8;
                v16bf a[4], b[2];
#pragma unroll
                for (int mb = 0; mb < 4; ++mb) a[mb] = load_frag(arow[mb] + kof);
#pragma unroll
                for (int j = 0; j < 2; ++j) {
                    b[j] = load_frag(bcol[j] + kof);
                    if (kc + 1 < D_IN / 32) __builtin_prefetch(bcol[j] + kof + 32);
                }
#pragma unroll
                for (int mb = 0; mb < 4; ++mb)
#pragma unroll
                    for (int j = 0; j < 2; ++j)
                        acc[mb][j] = __builtin_amdgcn_wmma_f32_16x16x32_bf16(
                            false, a[mb], false, b[j], (short)0, acc[mb][j],
                            false, false);
            }
#pragma unroll
            for (int j = 0; j < 2; ++j) {
                const int col  = (nbase + j) * 16 + lo;
                const float bias = b1[e * H_DIM + col];
#pragma unroll
                for (int mb = 0; mb < 4; ++mb)
#pragma unroll
                    for (int r = 0; r < 8; ++r) {
                        float v = acc[mb][j][r] + bias;
                        v = v > 0.0f ? v : 0.0f;
                        h1s[(mb * 16 + r + 8 * hi) * H_LD + col] = (__bf16)v;
                    }
            }
        }
    }
    __syncthreads();   // h1 ready, y_acc zeroed

    // ---- Phase B: h2 = relu(h1 @ W2[e] + b2[e]) in registers, folded into
    //      y_partial += h2 . w3[e]  (no LDS round-trip for h2) ----
    {
        const __bf16* w2e = W2p + (size_t)e * G_DIM * H_DIM;
        const float*  w3e = W3  + (size_t)e * G_DIM;
        const __bf16* h1row[4];
#pragma unroll
        for (int mb = 0; mb < 4; ++mb)
            h1row[mb] = h1s + (size_t)(mb * 16 + lo) * H_LD;

        float yp[4][8];
#pragma unroll
        for (int mb = 0; mb < 4; ++mb)
#pragma unroll
            for (int r = 0; r < 8; ++r) yp[mb][r] = 0.0f;

        for (int ng = 0; ng < 4; ++ng) {           // N-groups of 2 (8 cols/wave)
            const int nbase = wv * 8 + ng * 2;
            const __bf16* bcol[2];
#pragma unroll
            for (int j = 0; j < 2; ++j)
                bcol[j] = w2e + (size_t)((nbase + j) * 16 + lo) * H_DIM;

            v8f acc[4][2];
#pragma unroll
            for (int mb = 0; mb < 4; ++mb)
#pragma unroll
                for (int j = 0; j < 2; ++j) acc[mb][j] = (v8f){};

            for (int kc = 0; kc < H_DIM / 32; ++kc) {
                const int kof = kc * 32 + hi * 8;
                v16bf a[4], b[2];
#pragma unroll
                for (int mb = 0; mb < 4; ++mb)
                    a[mb] = load_frag(h1row[mb] + kof);        // ds_load_b128
#pragma unroll
                for (int j = 0; j < 2; ++j) {
                    b[j] = load_frag(bcol[j] + kof);
                    if (kc + 1 < H_DIM / 32) __builtin_prefetch(bcol[j] + kof + 32);
                }
#pragma unroll
                for (int mb = 0; mb < 4; ++mb)
#pragma unroll
                    for (int j = 0; j < 2; ++j)
                        acc[mb][j] = __builtin_amdgcn_wmma_f32_16x16x32_bf16(
                            false, a[mb], false, b[j], (short)0, acc[mb][j],
                            false, false);
            }
#pragma unroll
            for (int j = 0; j < 2; ++j) {
                const int col  = (nbase + j) * 16 + lo;
                const float bias = b2[e * G_DIM + col];
                const float w3c  = w3e[col];
#pragma unroll
                for (int mb = 0; mb < 4; ++mb)
#pragma unroll
                    for (int r = 0; r < 8; ++r) {
                        float v = acc[mb][j][r] + bias;
                        v = v > 0.0f ? v : 0.0f;
                        yp[mb][r] += v * w3c;                  // fold layer 3
                    }
            }
        }
        // reduce per-lane partials into y_acc (rows m = mb*16 + r + 8*hi)
#pragma unroll
        for (int mb = 0; mb < 4; ++mb)
#pragma unroll
            for (int r = 0; r < 8; ++r)
                atomicAdd(&y_acc[mb * 16 + r + 8 * hi], yp[mb][r]);
    }
    __syncthreads();

    // ---- Phase C: sigmoid + writeout ----
    if (threadIdx.x < MT) {
        float s = y_acc[threadIdx.x] + b3[e];
        ybuf[(size_t)e * N_TOK + m0 + threadIdx.x] = 1.0f / (1.0f + expf(-s));
    }
}

// --------------------------------------------------------------- combine
__global__ void combine_kernel(const float* __restrict__ g,
                               const float* __restrict__ ybuf,
                               float* __restrict__ pred) {
    int n = blockIdx.x * blockDim.x + threadIdx.x;
    if (n >= N_TOK) return;
    float s = 0.0f;
#pragma unroll
    for (int e = 0; e < N_EXP; ++e)
        s += g[n * N_EXP + e] * ybuf[(size_t)e * N_TOK + n];
    pred[n] = s;
}

// ---------------------------------------------------------------------------
extern "C" void kernel_launch(void* const* d_in, const int* in_sizes, int n_in,
                              void* d_out, int out_size, void* d_ws, size_t ws_size,
                              hipStream_t stream) {
    const float* x     = (const float*)d_in[0];   // [N,D]
    const float* gates = (const float*)d_in[1];   // [N,E]
    const float* W1    = (const float*)d_in[2];   // [E,D,H]
    const float* b1    = (const float*)d_in[3];   // [E,H]
    const float* W2    = (const float*)d_in[4];   // [E,H,G]
    const float* b2    = (const float*)d_in[5];   // [E,G]
    const float* W3    = (const float*)d_in[6];   // [E,G,1]
    const float* b3    = (const float*)d_in[7];   // [E,1]
    float* out = (float*)d_out;                   // [pred(4096) | g(4096*8)]

    char* ws = (char*)d_ws;
    const size_t XB_B  = (size_t)N_TOK * D_IN * 2;           //  8 MB
    const size_t W1P_B = (size_t)N_EXP * H_DIM * D_IN * 2;   // 32 MB
    const size_t W2P_B = (size_t)N_EXP * G_DIM * H_DIM * 2;  // 32 MB
    __bf16* Xb   = (__bf16*)ws;
    __bf16* W1p  = (__bf16*)(ws + XB_B);
    __bf16* W2p  = (__bf16*)(ws + XB_B + W1P_B);
    float*  ybuf = (float*)(ws + XB_B + W1P_B + W2P_B);      // [E,N]

    convert_x_kernel<<<(N_TOK * D_IN / 4) / 256, 256, 0, stream>>>(x, Xb);

    pack_transpose_kernel<<<dim3(H_DIM / 32, D_IN / 32, N_EXP), dim3(32, 8),
                            0, stream>>>(W1, W1p, D_IN, H_DIM);
    pack_transpose_kernel<<<dim3(G_DIM / 32, H_DIM / 32, N_EXP), dim3(32, 8),
                            0, stream>>>(W2, W2p, H_DIM, G_DIM);

    gating_kernel<<<N_TOK / 256, 256, 0, stream>>>(gates, out + N_TOK);

    fused_moe_kernel<<<dim3(N_TOK / MT, N_EXP), 256, SMEM_BYTES, stream>>>(
        Xb, W1p, W2p, b1, b2, W3, b3, ybuf);

    combine_kernel<<<N_TOK / 256, 256, 0, stream>>>(out + N_TOK, ybuf, out);
}
